// TransProModel_43035572306414
// MI455X (gfx1250) — matrix operand order repaired and checked by reading, not compiled, not run on hardware
//
#include <hip/hip_runtime.h>

typedef __attribute__((ext_vector_type(16))) _Float16 v16h;
typedef __attribute__((ext_vector_type(8)))  _Float16 v8h;
typedef __attribute__((ext_vector_type(8)))  float    v8f;
typedef __attribute__((ext_vector_type(4)))  float    v4f;

#define EMB 128
#define HID 256
#define BN_EPS 1e-5f

// ---------------- initial node embedding: sum of 6 categorical lookups ----------------
__global__ void k_init_embed(const int* __restrict__ x,
                             const float* __restrict__ ea, const float* __restrict__ ed,
                             const float* __restrict__ ec, const float* __restrict__ eh,
                             const float* __restrict__ er, const float* __restrict__ echir,
                             float* __restrict__ h, int N)
{
    int idx = blockIdx.x * blockDim.x + threadIdx.x;
    if (idx >= N * EMB) return;
    int n = idx >> 7, d = idx & 127;
    const int* xr = x + n * 6;
    h[idx] = ea[xr[0] * EMB + d] + ed[xr[1] * EMB + d] + ec[xr[2] * EMB + d]
           + eh[xr[3] * EMB + d] + er[xr[4] * EMB + d] + echir[xr[5] * EMB + d];
}

// ---------------- agg init with self-loop message: agg[n] = h[n] + ee1[4] + ee2[0] ----
__global__ void k_agg_init(const float* __restrict__ h, const float* __restrict__ ee1,
                           const float* __restrict__ ee2, float* __restrict__ agg, int N)
{
    int idx = blockIdx.x * blockDim.x + threadIdx.x;
    if (idx >= N * EMB) return;
    int d = idx & 127;
    agg[idx] = h[idx] + ee1[4 * EMB + d] + ee2[0 * EMB + d];
}

// ---------------- edge scatter: one wave per edge, 4 floats per lane ------------------
__global__ void k_scatter(const float* __restrict__ h, const float* __restrict__ ee1,
                          const float* __restrict__ ee2, const int* __restrict__ ei,
                          const int* __restrict__ ea, float* __restrict__ agg, int E)
{
    int e = blockIdx.x * 8 + (threadIdx.x >> 5);
    if (e >= E) return;
    int lane = threadIdx.x & 31;
    int s  = ei[e];
    int d  = ei[E + e];
    int a0 = ea[2 * e];
    int a1 = ea[2 * e + 1];
    int off = lane * 4;
    v4f hv = *(const v4f*)(h   + (long)s  * EMB + off);
    v4f e1 = *(const v4f*)(ee1 + (long)a0 * EMB + off);
    v4f e2 = *(const v4f*)(ee2 + (long)a1 * EMB + off);
    float* dp = agg + (long)d * EMB + off;
#pragma unroll
    for (int j = 0; j < 4; ++j)
        atomicAdd(dp + j, hv[j] + e1[j] + e2[j]);  // global_atomic_add_f32, L2-resident
}

// ---------------- elementwise f32 -> f16 (vectorized: 4 per thread) -------------------
__global__ void k_cvt_f16(const float* __restrict__ src, _Float16* __restrict__ dst,
                          int n4)
{
    int idx = blockIdx.x * blockDim.x + threadIdx.x;
    if (idx >= n4) return;
    v4f v = *(const v4f*)(src + idx * 4);
    _Float16* o = dst + idx * 4;
#pragma unroll
    for (int j = 0; j < 4; ++j) o[j] = (_Float16)v[j];
}

// ---------------- convert W[K,Nout] f32 -> Wt[Nout,K] f16 -----------------------------
__global__ void k_wt_convert(const float* __restrict__ W, _Float16* __restrict__ Wt,
                             int K, int Nout)
{
    int idx = blockIdx.x * blockDim.x + threadIdx.x;
    if (idx >= K * Nout) return;
    int n = idx / K, k = idx % K;
    Wt[idx] = (_Float16)W[k * Nout + n];
}

// ---------------- WMMA GEMM: C = act(A16[M,K] @ Bt[Nout,K]^T + bias) ------------------
// One wave computes one 16x16 output tile via v_wmma_f32_16x16x32_f16.
// A already f16 row-major; B stored transposed f16. Output: f16 (C16) or f32 (C32).
__global__ void k_gemm_wmma(const _Float16* __restrict__ A,
                            const _Float16* __restrict__ Bt,
                            const float* __restrict__ bias,
                            float* __restrict__ C32, _Float16* __restrict__ C16,
                            int Mrows, int K, int Nout, int relu, int out_f16)
{
    int wave   = blockIdx.x * (blockDim.x >> 5) + (threadIdx.x >> 5);
    int tilesN = Nout >> 4;
    int total  = (Mrows >> 4) * tilesN;
    if (wave >= total) return;                    // uniform per-wave -> EXEC all-1s
    int tm = wave / tilesN, tn = wave % tilesN;
    int lane = threadIdx.x & 31;
    int hsel = lane >> 4;
    int m = (tm << 4) + (lane & 15);
    int n = (tn << 4) + (lane & 15);

    float bv = bias[n];
    v8f c;
#pragma unroll
    for (int r = 0; r < 8; ++r) c[r] = bv;

    const _Float16* arow = A  + (long)m * K;
    const _Float16* brow = Bt + (long)n * K;

    for (int kc = 0; kc < K; kc += 32) {
        // A fragment: lane<16 -> row m, K kc..kc+15; lane>=16 -> K kc+16..kc+31
        int kb = kc + hsel * 16;
        v8h a0 = *(const v8h*)(arow + kb);        // 128-bit load
        v8h a1 = *(const v8h*)(arow + kb + 8);    // 128-bit load
        v16h a;
#pragma unroll
        for (int i = 0; i < 8; ++i) { a[i] = a0[i]; a[8 + i] = a1[i]; }
        // B fragment: lanes 0-15 hold K=kc..kc+15, lanes 16-31 hold K=kc+16..kc+31
        v16h b = *(const v16h*)(brow + kc + hsel * 16);
        c = __builtin_amdgcn_wmma_f32_16x16x32_f16(false, a, false, b,
                                                   (short)0, c, false, false);
    }

    int nb = (tn << 4) + (lane & 15);
    int mb = (tm << 4) + (hsel << 3);
#pragma unroll
    for (int r = 0; r < 8; ++r) {
        float v = c[r];
        if (relu) v = v > 0.f ? v : 0.f;
        if (out_f16) C16[(long)(mb + r) * Nout + nb] = (_Float16)v;
        else         C32[(long)(mb + r) * Nout + nb] = v;
    }
}

// ---------------- per-column sum / sumsq over N rows (for training-mode BN) -----------
__global__ void k_colstats(const float* __restrict__ hn, float* __restrict__ stats,
                           int N, int rowsPerBlock)
{
    int c  = threadIdx.x;                 // 128 threads = 128 columns
    int r0 = blockIdx.x * rowsPerBlock;
    int r1 = r0 + rowsPerBlock; if (r1 > N) r1 = N;
    float s = 0.f, s2 = 0.f;
    for (int r = r0; r < r1; ++r) {
        float v = hn[(long)r * EMB + c];
        s += v; s2 += v * v;
    }
    atomicAdd(&stats[c], s);
    atomicAdd(&stats[EMB + c], s2);
}

// ---------------- BN apply (+optional ReLU) -------------------------------------------
__global__ void k_bn(const float* __restrict__ hn, const float* __restrict__ stats,
                     const float* __restrict__ gamma, const float* __restrict__ beta,
                     float* __restrict__ out, int N, int relu)
{
    int idx = blockIdx.x * blockDim.x + threadIdx.x;
    if (idx >= N * EMB) return;
    int c = idx & 127;
    float inv_n = 1.f / (float)N;
    float mean  = stats[c] * inv_n;
    float var   = stats[EMB + c] * inv_n - mean * mean;   // biased var (jnp.var)
    float v = (hn[idx] - mean) * rsqrtf(var + BN_EPS) * gamma[c] + beta[c];
    if (relu) v = fmaxf(v, 0.f);
    out[idx] = v;
}

extern "C" void kernel_launch(void* const* d_in, const int* in_sizes, int n_in,
                              void* d_out, int out_size, void* d_ws, size_t ws_size,
                              hipStream_t stream)
{
    const int*   x         = (const int*)d_in[0];
    const int*   ei        = (const int*)d_in[1];   // [2,E] : row0=src, row1=dst
    const int*   ea        = (const int*)d_in[2];   // [E,2]
    const float* emb_atom  = (const float*)d_in[3];
    const float* emb_deg   = (const float*)d_in[4];
    const float* emb_chg   = (const float*)d_in[5];
    const float* emb_hyb   = (const float*)d_in[6];
    const float* emb_aro   = (const float*)d_in[7];
    const float* emb_chi   = (const float*)d_in[8];
    const float* edge_emb1 = (const float*)d_in[9];   // [L,6,128]
    const float* edge_emb2 = (const float*)d_in[10];  // [L,3,128]
    const float* W1        = (const float*)d_in[11];  // [L,128,256]
    const float* b1        = (const float*)d_in[12];
    const float* W2        = (const float*)d_in[13];  // [L,256,128]
    const float* b2        = (const float*)d_in[14];
    const float* bn_gamma  = (const float*)d_in[15];
    const float* bn_beta   = (const float*)d_in[16];

    const int N = in_sizes[0] / 6;                 // 50000 (multiple of 16)
    const int E = in_sizes[1] / 2;                 // 800000
    const int L = in_sizes[11] / (EMB * 2 * EMB);  // 3

    // workspace layout (all chunks 256B aligned)
    char*  ws   = (char*)d_ws;
    size_t szNE = (size_t)N * EMB * sizeof(float);       // 25.6 MB
    float*    h     = (float*)(ws);
    float*    agg   = (float*)(ws + szNE);
    float*    hn    = (float*)(ws + 2 * szNE);
    _Float16* a16   = (_Float16*)(ws + 3 * szNE);                          // N*128 f16
    _Float16* hid16 = (_Float16*)(ws + 3 * szNE + (size_t)N * EMB * 2);    // N*256 f16
    char*     tail  = ws + 3 * szNE + (size_t)N * EMB * 2 + (size_t)N * HID * 2;
    _Float16* W1t   = (_Float16*)(tail);
    _Float16* W2t   = (_Float16*)(tail + (size_t)EMB * HID * 2);
    float*    stats = (float*)(tail + (size_t)EMB * HID * 4);

    const int tot = N * EMB;
    k_init_embed<<<(tot + 255) / 256, 256, 0, stream>>>(
        x, emb_atom, emb_deg, emb_chg, emb_hyb, emb_aro, emb_chi, h, N);

    for (int l = 0; l < L; ++l) {
        const float* ee1 = edge_emb1 + (size_t)l * 6 * EMB;
        const float* ee2 = edge_emb2 + (size_t)l * 3 * EMB;

        k_wt_convert<<<(EMB * HID + 255) / 256, 256, 0, stream>>>(
            W1 + (size_t)l * EMB * HID, W1t, EMB, HID);
        k_wt_convert<<<(HID * EMB + 255) / 256, 256, 0, stream>>>(
            W2 + (size_t)l * HID * EMB, W2t, HID, EMB);
        hipMemsetAsync(stats, 0, 2 * EMB * sizeof(float), stream);

        k_agg_init<<<(tot + 255) / 256, 256, 0, stream>>>(h, ee1, ee2, agg, N);
        k_scatter<<<(E + 7) / 8, 256, 0, stream>>>(h, ee1, ee2, ei, ea, agg, E);

        // one-time f32 -> f16 conversion of the GEMM1 activation
        k_cvt_f16<<<(tot / 4 + 255) / 256, 256, 0, stream>>>(agg, a16, tot / 4);

        int tiles1 = (N / 16) * (HID / 16);
        k_gemm_wmma<<<(tiles1 + 7) / 8, 256, 0, stream>>>(
            a16, W1t, b1 + (size_t)l * HID, (float*)nullptr, hid16,
            N, EMB, HID, /*relu=*/1, /*out_f16=*/1);
        int tiles2 = (N / 16) * (EMB / 16);
        k_gemm_wmma<<<(tiles2 + 7) / 8, 256, 0, stream>>>(
            hid16, W2t, b2 + (size_t)l * EMB, hn, (_Float16*)nullptr,
            N, HID, EMB, /*relu=*/0, /*out_f16=*/0);

        k_colstats<<<(N + 511) / 512, 128, 0, stream>>>(hn, stats, N, 512);

        float* dest = (l == L - 1) ? (float*)d_out : h;
        k_bn<<<(tot + 255) / 256, 256, 0, stream>>>(
            hn, stats, bn_gamma + (size_t)l * EMB, bn_beta + (size_t)l * EMB,
            dest, N, (l < L - 1) ? 1 : 0);
    }
}
